// Model_996432413259
// MI455X (gfx1250) — matrix operand order repaired and checked by reading, not compiled, not run on hardware
//
#include <hip/hip_runtime.h>
#include <hip/hip_bf16.h>
#include <stdint.h>

// ---------------- CDNA5 WMMA types ----------------
typedef __attribute__((ext_vector_type(16))) __bf16 v16bf;
typedef __attribute__((ext_vector_type(8)))  float  v8f;

#define BATCH 256
#define SEQ   512
#define HID   512
#define INSZ  64

#define BT 64                 // batch rows per workgroup
#define JC 32                 // hidden columns per workgroup (gate cols = 4*JC = 128)
#define NSLICE (HID / JC)     // 16
#define NBT    (BATCH / BT)   // 4
#define NWG    (NSLICE * NBT) // 64 persistent workgroups (1 per WGP)
#define THREADS 256           // 8 wave32s
#define GP 132                // gates LDS row pitch in floats (128 + 4 pad)

union V16 { v16bf v; uint4 q[2]; };

__device__ __forceinline__ float sigf(float x)   { return 1.0f / (1.0f + __expf(-x)); }
__device__ __forceinline__ float tanhf_(float x) { return 2.0f / (1.0f + __expf(-2.0f * x)) - 1.0f; }

// ---------------- prep: pack [W_ih | W_hh] into bf16, sliced by hidden column group ----
// Layout: Wp[slice][lr][k], lr = gate*32 + c  (gate-major within slice), k in [0,K)
__global__ void pack_w_kernel(const float* __restrict__ Wih, const float* __restrict__ Whh,
                              __bf16* __restrict__ Wp, int IN, int K) {
  size_t idx = (size_t)blockIdx.x * blockDim.x + threadIdx.x;
  size_t total = (size_t)NSLICE * 128 * K;
  if (idx >= total) return;
  int k  = (int)(idx % K);
  int lr = (int)((idx / K) % 128);
  int s  = (int)(idx / ((size_t)K * 128));
  int g = lr >> 5, c = lr & 31;
  int grow = g * HID + s * JC + c;       // row in the 2048-row gate matrix
  float v = (k < IN) ? Wih[(size_t)grow * IN + k]
                     : Whh[(size_t)grow * HID + (k - IN)];
  Wp[idx] = (__bf16)v;
}

__global__ void cvt_x_kernel(const float* __restrict__ x, __bf16* __restrict__ xb, size_t n) {
  size_t i = (size_t)blockIdx.x * blockDim.x + threadIdx.x;
  if (i < n) xb[i] = (__bf16)x[i];
}

// ---------------- persistent LSTM layer (grid-synced recurrence) ----------------
// gates[b, 0:128] = [x_t(b) | h(b)] @ Wslice^T   via v_wmma_f32_16x16x32_bf16
template <int IN, bool L0>
__global__ __launch_bounds__(THREADS)
void lstm_layer_kernel(const __bf16* __restrict__ xin,   // L0: [b][t][64]; else [t][b][512]
                       const __bf16* __restrict__ Wp,    // [16][128][K] bf16
                       const float*  __restrict__ bih,
                       const float*  __restrict__ bhh,
                       __bf16* __restrict__ hbuf,        // double buffer: 2 * BATCH * HID
                       __bf16* __restrict__ hseq,        // layer0 output seq [t][b][j]
                       int* __restrict__ counters)       // SEQ counters, pre-zeroed
{
  constexpr int K  = IN + HID;
  constexpr int KP = K + 8;              // LDS row pitch in halves (bank-conflict pad)
  const int tid  = threadIdx.x;
  const int lane = tid & 31;
  const int wave = tid >> 5;
  const int btile = blockIdx.x & (NBT - 1);
  const int slice = blockIdx.x >> 2;

  extern __shared__ char smem[];
  __bf16* wlds = (__bf16*)smem;                                      // 128 rows x KP halves
  float*  glds = (float*)(smem + (size_t)128 * KP * sizeof(__bf16)); // BT x GP gates
  float*  cst  = glds + (size_t)BT * GP;                             // BT*JC cell state
  float*  bias = cst + BT * JC;                                      // 128 combined biases

  // --- pin the weight slice into LDS (stays resident for all 512 steps) ---
  constexpr int nch = K >> 3; // 16B chunks per row
  const __bf16* wsrc = Wp + (size_t)slice * 128 * K;
  for (int i = tid; i < 128 * nch; i += THREADS) {
    int row = i / nch, ch = i - row * nch;
    *(uint4*)(wlds + (size_t)row * KP + ch * 8) =
        *(const uint4*)(wsrc + (size_t)row * K + ch * 8);
  }
  if (tid < 128) {
    int g = tid >> 5, c = tid & 31;
    int grow = g * HID + slice * JC + c;
    bias[tid] = bih[grow] + bhh[grow];
  }
  for (int i = tid; i < BT * JC; i += THREADS) cst[i] = 0.0f;
  __syncthreads();

  // tile assignment: wave w -> M-tile (w&3), N-tiles 4*(w>>2) .. +3  (32 tiles total)
  const int mt    = wave & 3;
  const int nbase = (wave >> 2) * 4;
  const int mrow  = lane & 15;
  const int hh    = lane >> 4;
  const int b     = btile * BT + mt * 16 + mrow;   // global batch row for A fragments

  // per-lane LDS base for this wave's 4 B-fragment columns (halves offset folded in)
  const __bf16* wl[4];
#pragma unroll
  for (int j = 0; j < 4; ++j)
    wl[j] = wlds + (size_t)((nbase + j) * 16 + mrow) * KP + 8 * hh;

  for (int t = 0; t < SEQ; ++t) {
    const __bf16* hrd = hbuf + (size_t)(t & 1) * BATCH * HID;
    __bf16*       hwr = hbuf + (size_t)((t + 1) & 1) * BATCH * HID;
    const __bf16* xrow = L0 ? xin + ((size_t)b * SEQ + t) * INSZ
                            : xin + ((size_t)t * BATCH + b) * HID;
    const __bf16* hrow = hrd + (size_t)b * HID;

    v8f acc[4] = {};

    // ---- phase 1: x contribution (K in [0, IN)) — branchless addressing ----
#pragma unroll 2
    for (int k0 = 0; k0 < IN; k0 += 32) {
      V16 a;
      a.q[0] = *(const uint4*)(xrow + k0 + 8 * hh);
      a.q[1] = *(const uint4*)(xrow + k0 + 8 * hh + 16);
#pragma unroll
      for (int j = 0; j < 4; ++j) {
        V16 bf;
        bf.q[0] = *(const uint4*)(wl[j] + k0);
        bf.q[1] = *(const uint4*)(wl[j] + k0 + 16);
        acc[j] = __builtin_amdgcn_wmma_f32_16x16x32_bf16(
            false, a.v, false, bf.v, (short)0, acc[j], false, false);
      }
    }
    // ---- phase 2: h recurrence contribution (K in [IN, K)) ----
#pragma unroll 2
    for (int k0 = 0; k0 < HID; k0 += 32) {
      V16 a;
      a.q[0] = *(const uint4*)(hrow + k0 + 8 * hh);
      a.q[1] = *(const uint4*)(hrow + k0 + 8 * hh + 16);
#pragma unroll
      for (int j = 0; j < 4; ++j) {
        V16 bf;
        bf.q[0] = *(const uint4*)(wl[j] + IN + k0);
        bf.q[1] = *(const uint4*)(wl[j] + IN + k0 + 16);
        acc[j] = __builtin_amdgcn_wmma_f32_16x16x32_bf16(
            false, a.v, false, bf.v, (short)0, acc[j], false, false);
      }
    }

    // scatter accumulators (C/D layout: VGPR r -> row r + 8*half, col = lane%16)
#pragma unroll
    for (int j = 0; j < 4; ++j) {
#pragma unroll
      for (int r = 0; r < 8; ++r) {
        glds[(size_t)(mt * 16 + 8 * hh + r) * GP + (nbase + j) * 16 + mrow] = acc[j][r];
      }
    }
    __syncthreads();

    // fused LSTM cell update for this WG's 64x32 slab (c stays in LDS)
#pragma unroll
    for (int u = 0; u < (BT * JC) / THREADS; ++u) {
      int idx = tid + u * THREADS;
      int bl = idx >> 5, jl = idx & 31;
      const float* gr = glds + (size_t)bl * GP;
      float iv = sigf  (gr[jl]      + bias[jl]);
      float fv = sigf  (gr[32 + jl] + bias[32 + jl]);
      float gv = tanhf_(gr[64 + jl] + bias[64 + jl]);
      float ov = sigf  (gr[96 + jl] + bias[96 + jl]);
      float cv = fv * cst[idx] + iv * gv;
      cst[idx] = cv;
      float hv = ov * tanhf_(cv);
      int gb = btile * BT + bl;
      int gj = slice * JC + jl;
      hwr[(size_t)gb * HID + gj] = (__bf16)hv;
      if (L0) hseq[((size_t)t * BATCH + gb) * HID + gj] = (__bf16)hv;
    }

    // grid-wide step barrier: one fresh counter per step (graph-replay safe)
    __threadfence();
    __syncthreads();
    if (tid == 0) {
      int* c = counters + t;
      __hip_atomic_fetch_add(c, 1, __ATOMIC_ACQ_REL, __HIP_MEMORY_SCOPE_AGENT);
      while (__hip_atomic_load(c, __ATOMIC_ACQUIRE, __HIP_MEMORY_SCOPE_AGENT) < NWG) {
        __builtin_amdgcn_s_sleep(1);
      }
    }
    __syncthreads();
  }
}

// ---------------- head: out[b] = relu(h_last[b]) . W_out + b_out ----------------
__global__ void head_kernel(const __bf16* __restrict__ hlast, const float* __restrict__ Wout,
                            const float* __restrict__ bout, float* __restrict__ out) {
  int b = blockIdx.x * blockDim.x + threadIdx.x;
  if (b >= BATCH) return;
  const __bf16* hr = hlast + (size_t)b * HID;
  float s = 0.f;
  for (int j = 0; j < HID; ++j) {
    float h = (float)hr[j];
    s += (h > 0.f ? h : 0.f) * Wout[j];
  }
  out[b] = s + bout[0];
}

// ---------------- launch ----------------
extern "C" void kernel_launch(void* const* d_in, const int* in_sizes, int n_in,
                              void* d_out, int out_size, void* d_ws, size_t ws_size,
                              hipStream_t stream) {
  (void)in_sizes; (void)n_in; (void)out_size; (void)ws_size;
  const float* x    = (const float*)d_in[0];
  const float* Wih0 = (const float*)d_in[1];
  const float* Whh0 = (const float*)d_in[2];
  const float* bih0 = (const float*)d_in[3];
  const float* bhh0 = (const float*)d_in[4];
  const float* Wih1 = (const float*)d_in[5];
  const float* Whh1 = (const float*)d_in[6];
  const float* bih1 = (const float*)d_in[7];
  const float* bhh1 = (const float*)d_in[8];
  const float* Wout = (const float*)d_in[9];
  const float* bout = (const float*)d_in[10];
  float* out = (float*)d_out;

  const int K0 = INSZ + HID, KP0 = K0 + 8;   // 576 / 584
  const int K1 = HID + HID,  KP1 = K1 + 8;   // 1024 / 1032

  char* ws = (char*)d_ws;
  size_t off = 0;
  int*    counters = (int*)(ws + off);   off += 2 * SEQ * sizeof(int);
  __bf16* hbufA    = (__bf16*)(ws + off); off += (size_t)2 * BATCH * HID * 2;
  __bf16* hbufB    = (__bf16*)(ws + off); off += (size_t)2 * BATCH * HID * 2;
  size_t zero_bytes = off;                               // counters + both h double-buffers
  __bf16* xb   = (__bf16*)(ws + off); off += (size_t)BATCH * SEQ * INSZ * 2;
  __bf16* Wp0  = (__bf16*)(ws + off); off += (size_t)NSLICE * 128 * K0 * 2;
  __bf16* Wp1  = (__bf16*)(ws + off); off += (size_t)NSLICE * 128 * K1 * 2;
  __bf16* hseq = (__bf16*)(ws + off); off += (size_t)SEQ * BATCH * HID * 2;

  hipMemsetAsync(ws, 0, zero_bytes, stream);

  size_t tot0 = (size_t)NSLICE * 128 * K0;
  pack_w_kernel<<<(unsigned)((tot0 + 255) / 256), 256, 0, stream>>>(Wih0, Whh0, Wp0, INSZ, K0);
  size_t tot1 = (size_t)NSLICE * 128 * K1;
  pack_w_kernel<<<(unsigned)((tot1 + 255) / 256), 256, 0, stream>>>(Wih1, Whh1, Wp1, HID, K1);
  size_t nx = (size_t)BATCH * SEQ * INSZ;
  cvt_x_kernel<<<(unsigned)((nx + 255) / 256), 256, 0, stream>>>(x, xb, nx);

  size_t sh0 = (size_t)128 * KP0 * 2 + (size_t)BT * GP * 4 + (size_t)BT * JC * 4 + 128 * 4;
  size_t sh1 = (size_t)128 * KP1 * 2 + (size_t)BT * GP * 4 + (size_t)BT * JC * 4 + 128 * 4;
  hipFuncSetAttribute(reinterpret_cast<const void*>(lstm_layer_kernel<INSZ, true>),
                      hipFuncAttributeMaxDynamicSharedMemorySize, (int)sh0);
  hipFuncSetAttribute(reinterpret_cast<const void*>(lstm_layer_kernel<HID, false>),
                      hipFuncAttributeMaxDynamicSharedMemorySize, (int)sh1);

  lstm_layer_kernel<INSZ, true ><<<NWG, THREADS, sh0, stream>>>(xb,   Wp0, bih0, bhh0,
                                                                hbufA, hseq, counters);
  lstm_layer_kernel<HID,  false><<<NWG, THREADS, sh1, stream>>>(hseq, Wp1, bih1, bhh1,
                                                                hbufB, nullptr, counters + SEQ);

  // after 512 steps, the final h sits in parity-0 half of hbufB
  head_kernel<<<(BATCH + 63) / 64, 64, 0, stream>>>(hbufB, Wout, bout, out);
}
